// LlamaUniQLAttention_83416854823028
// MI455X (gfx1250) — matrix operand order repaired and checked by reading, compiled, not run on hardware
//
#include <hip/hip_runtime.h>
#include <hip/hip_bf16.h>

// ---------------------------------------------------------------------------
// Problem constants (from reference)
// ---------------------------------------------------------------------------
#define S_LEN   2048
#define HID     4096
#define NH      32          // query heads
#define NKV     8           // kv heads
#define DH      96          // head dim
#define DFULL   128         // rope table dim
#define QKV_N   (NH*DH + 2*NKV*DH)   // 4608
#define OPROJ_K (NH*DH)              // 3072
#define SCALE_F 0.08838834764831843f // 128^-0.5 (reference uses DFULL)

typedef __bf16  v16bf __attribute__((ext_vector_type(16)));
typedef __bf16  v8bf  __attribute__((ext_vector_type(8)));
typedef float   v8f   __attribute__((ext_vector_type(8)));
typedef int     v4i   __attribute__((ext_vector_type(4)));

#if __has_builtin(__builtin_amdgcn_global_load_async_to_lds_b128) && \
    __has_builtin(__builtin_amdgcn_s_wait_asynccnt)
#define USE_ASYNC_LDS 1
#else
#define USE_ASYNC_LDS 0
#endif

__device__ inline v8f wmma_bf16(v16bf a, v16bf b, v8f c) {
  return __builtin_amdgcn_wmma_f32_16x16x32_bf16(false, a, false, b, (short)0, c,
                                                 false, false);
}

__device__ inline v8f zero_v8f() {
  v8f z;
#pragma unroll
  for (int i = 0; i < 8; ++i) z[i] = 0.0f;
  return z;
}

// A-fragment (16x32 MxK, 16-bit): lane half h holds K in [16h,16h+16).
// base points at row start + k-tile base (32B-aligned chunks).
__device__ inline v16bf load_a_frag(const __bf16* base, int h) {
  return *(const v16bf*)(base + 16 * h);
}

// ---------------------------------------------------------------------------
// fp32 -> bf16 convert (grid-stride)
// ---------------------------------------------------------------------------
__global__ void f2bf_kernel(const float* __restrict__ in,
                            __bf16* __restrict__ out, size_t n) {
  size_t i = (size_t)blockIdx.x * blockDim.x + threadIdx.x;
  size_t stride = (size_t)gridDim.x * blockDim.x;
  for (; i < n; i += stride) out[i] = (__bf16)in[i];
}

// ---------------------------------------------------------------------------
// LDS-tiled GEMM: C[M,N] (fp32) = A[M,K] * B[N,K]^T, A/B bf16 row-major.
// Block: 256 threads (8 waves) -> 128x128 C tile; wave grid 4(M) x 2(N),
// wave tile 32x64.  K-step 32, LDS tiles double-buffered, async-to-LDS when
// the toolchain exposes the gfx1250 builtin (ASYNCcnt tracked), else a
// synchronous VGPR->LDS fallback.  LDS row stride padded to 48 elems (96B).
// grid = (N/128, M/128).
// ---------------------------------------------------------------------------
#define BK      32
#define LDS_STR 48

__global__ __launch_bounds__(256)
void gemm_lds_kernel(const __bf16* __restrict__ A, const __bf16* __restrict__ B,
                     float* __restrict__ C, int K, int ldc) {
  __shared__ __align__(32) __bf16 As[2][128 * LDS_STR];
  __shared__ __align__(32) __bf16 Bs[2][128 * LDS_STR];

  const int tid = threadIdx.x;
  const int lane = tid & 31, lm = lane & 15, hf = lane >> 4;
  const int wave = tid >> 5, mw = wave >> 1, nw = wave & 1;
  const int mb = blockIdx.y * 128, nb = blockIdx.x * 128;

  // global->LDS assignment: each thread owns a 32B chunk (row, half-row)
  const int arow = tid >> 1, achk = tid & 1;
  const __bf16* ag = A + (size_t)(mb + arow) * K + achk * 16;
  const __bf16* bg = B + (size_t)(nb + arow) * K + achk * 16;
  const int lds_off = arow * LDS_STR + achk * 16;

  v8f acc[2][4];
#pragma unroll
  for (int i = 0; i < 2; ++i)
#pragma unroll
    for (int j = 0; j < 4; ++j) acc[i][j] = zero_v8f();

  const int nk = K / BK;

  auto compute = [&](int b) {
    const __bf16* as = &As[b][0];
    const __bf16* bs = &Bs[b][0];
    v16bf af[2], bf[4];
#pragma unroll
    for (int mi = 0; mi < 2; ++mi)
      af[mi] = load_a_frag(&as[(mw * 32 + mi * 16 + lm) * LDS_STR], hf);
#pragma unroll
    for (int j = 0; j < 4; ++j)
      bf[j] = *(const v16bf*)(&bs[(nw * 64 + j * 16 + lm) * LDS_STR + 16 * hf]);
#pragma unroll
    for (int mi = 0; mi < 2; ++mi)
#pragma unroll
      for (int j = 0; j < 4; ++j)
        acc[mi][j] = wmma_bf16(af[mi], bf[j], acc[mi][j]);
  };

#if USE_ASYNC_LDS
  auto issue = [&](int kt, int b) {
    const __bf16* gA = ag + (size_t)kt * BK;
    const __bf16* gB = bg + (size_t)kt * BK;
    __bf16* lA = &As[b][lds_off];
    __bf16* lB = &Bs[b][lds_off];
    __builtin_amdgcn_global_load_async_to_lds_b128((v4i*)gA, (v4i*)lA, 0, 0);
    __builtin_amdgcn_global_load_async_to_lds_b128((v4i*)gA, (v4i*)lA, 16, 0);
    __builtin_amdgcn_global_load_async_to_lds_b128((v4i*)gB, (v4i*)lB, 0, 0);
    __builtin_amdgcn_global_load_async_to_lds_b128((v4i*)gB, (v4i*)lB, 16, 0);
  };
  issue(0, 0);
  for (int k = 0; k < nk - 1; ++k) {
    issue(k + 1, (k + 1) & 1);                 // prefetch next tile
    __builtin_amdgcn_s_wait_asynccnt(4);       // tile k landed (in-order)
    __syncthreads();
    compute(k & 1);
    __syncthreads();                           // computed buffer free for reuse
  }
  __builtin_amdgcn_s_wait_asynccnt(0);
  __syncthreads();
  compute((nk - 1) & 1);
#else
  for (int k = 0; k < nk; ++k) {
    v16bf av = *(const v16bf*)(ag + (size_t)k * BK);
    v16bf bv = *(const v16bf*)(bg + (size_t)k * BK);
    __syncthreads();                           // previous compute done reading
    *(v16bf*)(&As[0][lds_off]) = av;
    *(v16bf*)(&Bs[0][lds_off]) = bv;
    __syncthreads();
    compute(0);
  }
#endif

#pragma unroll
  for (int mi = 0; mi < 2; ++mi)
#pragma unroll
    for (int j = 0; j < 4; ++j)
#pragma unroll
      for (int r = 0; r < 8; ++r)
        C[(size_t)(mb + mw * 32 + mi * 16 + r + 8 * hf) * ldc +
          nb + nw * 64 + j * 16 + lm] = acc[mi][j][r];
}

// ---------------------------------------------------------------------------
// RoPE + repack.  QKV fp32 [S, 4608] -> Qb [NH][S][DH] bf16 (roped),
// Kb [NKV][S][DH] bf16 (roped), Vt [NKV][DH][S] bf16 (transposed).
// ---------------------------------------------------------------------------
__global__ void rope_kernel(const float* __restrict__ QKV,
                            const float* __restrict__ cosT,
                            const float* __restrict__ sinT,
                            const int* __restrict__ idx,
                            __bf16* __restrict__ Qb, __bf16* __restrict__ Kb,
                            __bf16* __restrict__ Vt) {
  long t = (long)blockIdx.x * blockDim.x + threadIdx.x;
  const long total = (long)S_LEN * 48 * DH;
  if (t >= total) return;
  int d = (int)(t % DH);
  long rest = t / DH;
  int slot = (int)(rest % 48);
  int s = (int)(rest / 48);
  const float* row = QKV + (size_t)s * QKV_N;
  int dp = (d < 48) ? d + 48 : d - 48;
  if (slot < 32) {
    int hq = slot, kvh = hq >> 2;
    float v = row[hq * DH + d];
    float pr = row[hq * DH + dp];
    int fi = idx[kvh * DH + d];
    float c = cosT[(size_t)s * DFULL + fi];
    float sn = sinT[(size_t)s * DFULL + fi];
    float o = v * c + ((d < 48) ? -pr : pr) * sn;
    Qb[((size_t)hq * S_LEN + s) * DH + d] = (__bf16)o;
  } else if (slot < 40) {
    int kvh = slot - 32;
    float v = row[NH * DH + kvh * DH + d];
    float pr = row[NH * DH + kvh * DH + dp];
    int fi = idx[kvh * DH + d];
    float c = cosT[(size_t)s * DFULL + fi];
    float sn = sinT[(size_t)s * DFULL + fi];
    float o = v * c + ((d < 48) ? -pr : pr) * sn;
    Kb[((size_t)kvh * S_LEN + s) * DH + d] = (__bf16)o;
  } else {
    int kvh = slot - 40;
    float v = row[NH * DH + NKV * DH + kvh * DH + d];
    Vt[((size_t)kvh * DH + d) * S_LEN + s] = (__bf16)v;
  }
}

// ---------------------------------------------------------------------------
// Causal GQA flash attention. One wave per (head, 16-query block).
// Key blocks of 32; Q.K^T = 6 WMMAs, P.V = 6 WMMAs per block.
// Output: Attn [S][NH*DH] bf16 (A operand of O-proj GEMM).
// ---------------------------------------------------------------------------
__global__ __launch_bounds__(32)
void attn_kernel(const __bf16* __restrict__ Qb, const __bf16* __restrict__ Kb,
                 const __bf16* __restrict__ Vt, __bf16* __restrict__ Attn) {
  const int h = blockIdx.x;          // 0..31
  const int qb = blockIdx.y;         // 0..127
  const int kvh = h >> 2;
  const int qbase = qb * 16;
  const int lane = threadIdx.x, lm = lane & 15, hf = lane >> 4;

  __shared__ __align__(32) __bf16 Plds[16 * 32];

  const __bf16* qrow = Qb + ((size_t)h * S_LEN + qbase + lm) * DH;
  v16bf aq[3];
#pragma unroll
  for (int t = 0; t < 3; ++t) aq[t] = load_a_frag(qrow + t * 32, hf);

  v8f acc[6];
#pragma unroll
  for (int j = 0; j < 6; ++j) acc[j] = zero_v8f();
  float mrun[8], lrun[8];
#pragma unroll
  for (int r = 0; r < 8; ++r) { mrun[r] = -1e30f; lrun[r] = 0.0f; }

  const int nblocks = (qbase >> 5) + 1;
  for (int kb = 0; kb < nblocks; ++kb) {
    const int kbase = kb * 32;
    v8f s0 = zero_v8f(), s1 = zero_v8f();
#pragma unroll
    for (int t = 0; t < 3; ++t) {
      v16bf b0 = *(const v16bf*)(Kb + ((size_t)kvh * S_LEN + kbase + lm) * DH +
                                 t * 32 + 16 * hf);
      v16bf b1 = *(const v16bf*)(Kb + ((size_t)kvh * S_LEN + kbase + 16 + lm) * DH +
                                 t * 32 + 16 * hf);
      s0 = wmma_bf16(aq[t], b0, s0);
      s1 = wmma_bf16(aq[t], b1, s1);
    }
    float p0[8], p1[8], alpha[8];
#pragma unroll
    for (int r = 0; r < 8; ++r) {
      const int row = qbase + r + 8 * hf;
      float v0 = s0[r] * SCALE_F;
      float v1 = s1[r] * SCALE_F;
      if (kbase + lm > row) v0 = -1e30f;          // causal mask
      if (kbase + 16 + lm > row) v1 = -1e30f;
      float m = fmaxf(v0, v1);
      m = fmaxf(m, __shfl_xor(m, 1, 32));
      m = fmaxf(m, __shfl_xor(m, 2, 32));
      m = fmaxf(m, __shfl_xor(m, 4, 32));
      m = fmaxf(m, __shfl_xor(m, 8, 32));
      float mnew = fmaxf(mrun[r], m);
      alpha[r] = __expf(mrun[r] - mnew);
      mrun[r] = mnew;
      float e0 = __expf(v0 - mnew);
      float e1 = __expf(v1 - mnew);
      p0[r] = e0; p1[r] = e1;
      float rs = e0 + e1;
      rs += __shfl_xor(rs, 1, 32);
      rs += __shfl_xor(rs, 2, 32);
      rs += __shfl_xor(rs, 4, 32);
      rs += __shfl_xor(rs, 8, 32);
      lrun[r] = lrun[r] * alpha[r] + rs;
    }
#pragma unroll
    for (int j = 0; j < 6; ++j)
#pragma unroll
      for (int r = 0; r < 8; ++r) acc[j][r] *= alpha[r];

    // C-layout (lane=n, reg=m) -> A-layout (lane=m, regs=k) via LDS
    __syncthreads();
#pragma unroll
    for (int r = 0; r < 8; ++r) {
      Plds[(r + 8 * hf) * 32 + lm] = (__bf16)p0[r];
      Plds[(r + 8 * hf) * 32 + 16 + lm] = (__bf16)p1[r];
    }
    __syncthreads();
    v16bf ap = load_a_frag(&Plds[lm * 32], hf);
#pragma unroll
    for (int j = 0; j < 6; ++j) {
      v16bf bv = *(const v16bf*)(Vt + ((size_t)kvh * DH + j * 16 + lm) * S_LEN +
                                 kbase + 16 * hf);
      acc[j] = wmma_bf16(ap, bv, acc[j]);
    }
  }
#pragma unroll
  for (int j = 0; j < 6; ++j)
#pragma unroll
    for (int r = 0; r < 8; ++r) {
      float o = acc[j][r] / lrun[r];
      Attn[(size_t)(qbase + r + 8 * hf) * OPROJ_K + h * DH + j * 16 + lm] =
          (__bf16)o;
    }
}

// ---------------------------------------------------------------------------
// Launch
// ---------------------------------------------------------------------------
extern "C" void kernel_launch(void* const* d_in, const int* in_sizes, int n_in,
                              void* d_out, int out_size, void* d_ws, size_t ws_size,
                              hipStream_t stream) {
  const float* X    = (const float*)d_in[0];   // [1,S,HID]
  const float* cosT = (const float*)d_in[1];   // [1,S,128]
  const float* sinT = (const float*)d_in[2];   // [1,S,128]
  const int*   idx  = (const int*)d_in[3];     // [NKV,96]
  const float* Wq   = (const float*)d_in[4];   // [3072,4096]
  const float* Wk   = (const float*)d_in[5];   // [768,4096]
  const float* Wv   = (const float*)d_in[6];   // [768,4096]
  const float* Wo   = (const float*)d_in[7];   // [4096,3072]
  float* out = (float*)d_out;                  // [1,S,HID] fp32

  char* ws = (char*)d_ws;
  const size_t OFF_XB   = 0;                                        // S*HID bf16
  const size_t OFF_WB   = OFF_XB  + (size_t)S_LEN * HID * 2;        // QKV_N*HID bf16
  const size_t OFF_WOB  = OFF_WB  + (size_t)QKV_N * HID * 2;        // HID*OPROJ_K bf16
  const size_t OFF_QKV  = OFF_WOB + (size_t)HID * OPROJ_K * 2;      // S*QKV_N f32
  const size_t OFF_QB   = OFF_QKV + (size_t)S_LEN * QKV_N * 4;      // NH*S*DH bf16
  const size_t OFF_KB   = OFF_QB  + (size_t)NH * S_LEN * DH * 2;    // NKV*S*DH bf16
  const size_t OFF_VT   = OFF_KB  + (size_t)NKV * S_LEN * DH * 2;   // NKV*DH*S bf16
  const size_t OFF_ATTN = OFF_VT  + (size_t)NKV * DH * S_LEN * 2;   // S*OPROJ_K bf16

  __bf16* Xb   = (__bf16*)(ws + OFF_XB);
  __bf16* Wb   = (__bf16*)(ws + OFF_WB);    // [Wq;Wk;Wv] as [4608][4096]
  __bf16* Wob  = (__bf16*)(ws + OFF_WOB);
  float*  QKVf = (float*)(ws + OFF_QKV);
  __bf16* Qb   = (__bf16*)(ws + OFF_QB);
  __bf16* Kb   = (__bf16*)(ws + OFF_KB);
  __bf16* Vt   = (__bf16*)(ws + OFF_VT);
  __bf16* Attn = (__bf16*)(ws + OFF_ATTN);

  // 1) convert everything to bf16
  f2bf_kernel<<<8192, 256, 0, stream>>>(X, Xb, (size_t)S_LEN * HID);
  f2bf_kernel<<<8192, 256, 0, stream>>>(Wq, Wb, (size_t)NH * DH * HID);
  f2bf_kernel<<<8192, 256, 0, stream>>>(Wk, Wb + (size_t)NH * DH * HID,
                                        (size_t)NKV * DH * HID);
  f2bf_kernel<<<8192, 256, 0, stream>>>(Wv, Wb + (size_t)(NH + NKV) * DH * HID,
                                        (size_t)NKV * DH * HID);
  f2bf_kernel<<<8192, 256, 0, stream>>>(Wo, Wob, (size_t)HID * OPROJ_K);

  // 2) QKV projection: [S,4608] = Xb[S,4096] . Wb[4608,4096]^T
  gemm_lds_kernel<<<dim3(QKV_N / 128, S_LEN / 128), 256, 0, stream>>>(
      Xb, Wb, QKVf, HID, QKV_N);

  // 3) indexed RoPE + repack (Q/K roped bf16, V transposed bf16)
  {
    long total = (long)S_LEN * 48 * DH;
    int blocks = (int)((total + 255) / 256);
    rope_kernel<<<blocks, 256, 0, stream>>>(QKVf, cosT, sinT, idx, Qb, Kb, Vt);
  }

  // 4) causal GQA flash attention
  attn_kernel<<<dim3(NH, S_LEN / 16), 32, 0, stream>>>(Qb, Kb, Vt, Attn);

  // 5) output projection: out[S,4096] = Attn[S,3072] . Wob[4096,3072]^T
  gemm_lds_kernel<<<dim3(HID / 128, S_LEN / 128), 256, 0, stream>>>(
      Attn, Wob, out, OPROJ_K, HID);
}
